// GraphStateFunction_13666585936303
// MI455X (gfx1250) — compile-verified, matrix operand order checked by reading, not run on hardware
//
#include <hip/hip_runtime.h>
#include <stdint.h>

// ---------------------------------------------------------------------------
// CDNA5 (gfx1250) bf16 WMMA GEMM pipeline for the HGNN reference.
//   D = act(A @ B + bias).  G pre-converted to bf16 once.  Heavy conv GEMMs:
//   A-tile staged with GLOBAL_LOAD_ASYNC_TO_LDS_B128 (ASYNCcnt), LDS double-
//   buffered, one barrier per K-step, 16 v_wmma_f32_16x16x32_bf16 per stage.
// Tiling: WG 256x128 (8 wave32, 4(M)x2(N)), wave 64x64, BLK_K = 32.
// ---------------------------------------------------------------------------

typedef __attribute__((ext_vector_type(16))) __bf16 v16bf;
typedef __attribute__((ext_vector_type(8)))  float  v8f;
typedef __attribute__((ext_vector_type(4)))  int    v4i;

#define BLK_M 256
#define BLK_N 128
#define BLK_K 32
#define LDSS  40                 // 32 + 8 pad elems -> 80B stride (16B aligned)
#define ASZ   (BLK_M * LDSS)     // one A buffer, in elements
#define BSZ   (BLK_N * LDSS)     // one B buffer, in elements

#if __has_builtin(__builtin_amdgcn_global_load_async_to_lds_b128)
#define HAS_ASYNC_LDS 1
#else
#define HAS_ASYNC_LDS 0
#endif

__device__ __forceinline__ void async_copy_b128(const void* g, void* l) {
#if HAS_ASYNC_LDS
  __builtin_amdgcn_global_load_async_to_lds_b128(
      (__attribute__((address_space(1))) v4i*)g,
      (__attribute__((address_space(3))) v4i*)l, 0, 0);
#else
  *(uint4*)l = *(const uint4*)g;
#endif
}
__device__ __forceinline__ void wait_async_lds() {
#if HAS_ASYNC_LDS
#if __has_builtin(__builtin_amdgcn_s_wait_asynccnt)
  __builtin_amdgcn_s_wait_asynccnt(0);
#else
  asm volatile("s_wait_asynccnt 0" ::: "memory");
#endif
#endif
}

__device__ __forceinline__ unsigned f2bfu(float f) {
  union { float f; unsigned u; } x; x.f = f;
  unsigned u = x.u;
  unsigned r = u + 0x7FFFu + ((u >> 16) & 1u);   // round-to-nearest-even
  return r >> 16;                                 // bf16 in low 16 bits
}
__device__ __forceinline__ float bf2f(unsigned short b) {
  union { unsigned u; float f; } x; x.u = ((unsigned)b) << 16; return x.f;
}

union FragBF { uint4 u4[2]; v16bf v; };

// ---- elementwise f32 -> bf16 (8 elements / thread, b128 in & out) ----
__global__ __launch_bounds__(256)
void f32_to_bf16_kernel(const float* __restrict__ src,
                        unsigned short* __restrict__ dst)
{
  const size_t i = (size_t)blockIdx.x * 256 + threadIdx.x;
  const float4* s = (const float4*)src + i * 2;
  float4 f0 = s[0];
  float4 f1 = s[1];
  uint4 p;
  p.x = f2bfu(f0.x) | (f2bfu(f0.y) << 16);
  p.y = f2bfu(f0.z) | (f2bfu(f0.w) << 16);
  p.z = f2bfu(f1.x) | (f2bfu(f1.y) << 16);
  p.w = f2bfu(f1.z) | (f2bfu(f1.w) << 16);
  ((uint4*)dst)[i] = p;
}

template<bool A_BF16, bool B_BF16, bool RELU>
__global__ __launch_bounds__(256)
void gemm_bias_act_bf16out(const void* __restrict__ Av, int lda, size_t aBatch,
                           const void* __restrict__ Bv, int ldb, size_t bBatch,
                           const float* __restrict__ bias,
                           unsigned short* __restrict__ D, int ldd, size_t dBatch,
                           int Ntot, int K)
{
  __shared__ __align__(16) unsigned short As[2 * ASZ];
  __shared__ __align__(16) unsigned short Bs[2 * BSZ];

  const int tid  = threadIdx.x;
  const int lane = tid & 31;
  const int wid  = tid >> 5;              // 0..7 (8 wave32s)
  const int wm   = (wid & 3) * 64;        // wave M offset inside WG tile
  const int wn   = (wid >> 2) * 64;       // wave N offset inside WG tile
  const int half = lane >> 4;             // 0 or 1
  const int l16  = lane & 15;

  const size_t z = blockIdx.z;
  const int blockM = blockIdx.x * BLK_M;
  const int blockN = blockIdx.y * BLK_N;

  const int ar = tid;                     // A: one thread per row (256 x 32)
  const int bk = (tid & 7) * 4;           // B: 4(K) x 4(N) block per thread
  const int bn = (tid >> 3) * 4;

  // ---- staging helpers ----
  auto stageA = [&](int k0, unsigned short* AsD) {
    if (A_BF16) {
      const unsigned short* src = (const unsigned short*)Av + aBatch * z +
                                  (size_t)(blockM + ar) * lda + k0;
#pragma unroll
      for (int q = 0; q < 4; ++q)
        async_copy_b128(src + q * 8, &AsD[ar * LDSS + q * 8]);
    } else {
      const float* src = (const float*)Av + aBatch * z +
                         (size_t)(blockM + ar) * lda + k0;
      uint4* dst = (uint4*)&AsD[ar * LDSS];
#pragma unroll
      for (int q = 0; q < 4; ++q) {
        float4 f0 = ((const float4*)src)[2 * q];
        float4 f1 = ((const float4*)src)[2 * q + 1];
        uint4 p;
        p.x = f2bfu(f0.x) | (f2bfu(f0.y) << 16);
        p.y = f2bfu(f0.z) | (f2bfu(f0.w) << 16);
        p.z = f2bfu(f1.x) | (f2bfu(f1.y) << 16);
        p.w = f2bfu(f1.z) | (f2bfu(f1.w) << 16);
        dst[q] = p;
      }
    }
  };

  auto stageB = [&](int k0, unsigned short* BsD) {
    const int gn = blockN + bn;
    const bool nok = (gn < Ntot);
    uint2 row[4];                         // [k] : 4 bf16 (n..n+3) each
#pragma unroll
    for (int r = 0; r < 4; ++r) {
      const size_t gk = (size_t)(k0 + bk + r);
      if (B_BF16) {
        uint2 u; u.x = 0u; u.y = 0u;
        if (nok) u = *(const uint2*)((const unsigned short*)Bv + bBatch * z + gk * ldb + gn);
        row[r] = u;
      } else {
        float4 f; f.x = f.y = f.z = f.w = 0.0f;
        if (nok) f = *(const float4*)((const float*)Bv + bBatch * z + gk * ldb + gn);
        uint2 u;
        u.x = f2bfu(f.x) | (f2bfu(f.y) << 16);
        u.y = f2bfu(f.z) | (f2bfu(f.w) << 16);
        row[r] = u;
      }
    }
    // 4x4 bf16 transpose with 32-bit ops
    uint2 c0, c1, c2, c3;
    c0.x = (row[0].x & 0xFFFFu)  | (row[1].x << 16);
    c0.y = (row[2].x & 0xFFFFu)  | (row[3].x << 16);
    c1.x = (row[0].x >> 16)      | (row[1].x & 0xFFFF0000u);
    c1.y = (row[2].x >> 16)      | (row[3].x & 0xFFFF0000u);
    c2.x = (row[0].y & 0xFFFFu)  | (row[1].y << 16);
    c2.y = (row[2].y & 0xFFFFu)  | (row[3].y << 16);
    c3.x = (row[0].y >> 16)      | (row[1].y & 0xFFFF0000u);
    c3.y = (row[2].y >> 16)      | (row[3].y & 0xFFFF0000u);
    *(uint2*)&BsD[(bn + 0) * LDSS + bk] = c0;
    *(uint2*)&BsD[(bn + 1) * LDSS + bk] = c1;
    *(uint2*)&BsD[(bn + 2) * LDSS + bk] = c2;
    *(uint2*)&BsD[(bn + 3) * LDSS + bk] = c3;
  };

  v8f acc[4][4];
#pragma unroll
  for (int i = 0; i < 4; ++i)
#pragma unroll
    for (int j = 0; j < 4; ++j)
#pragma unroll
      for (int r = 0; r < 8; ++r) acc[i][j][r] = 0.0f;

  // ---- software-pipelined main loop (double-buffered LDS) ----
  stageA(0, As);
  stageB(0, Bs);
  wait_async_lds();
  __syncthreads();

  int cur = 0;
  for (int k0 = 0; k0 < K; k0 += BLK_K) {
    const int nxt = cur ^ 1;
    const bool more = (k0 + BLK_K) < K;
    if (more) {               // overlap next-tile staging with this compute
      stageA(k0 + BLK_K, As + nxt * ASZ);
      stageB(k0 + BLK_K, Bs + nxt * BSZ);
    }

    const unsigned short* Asb = As + cur * ASZ;
    const unsigned short* Bsb = Bs + cur * BSZ;
    FragBF a[4], b[4];
#pragma unroll
    for (int i = 0; i < 4; ++i) {
      const unsigned short* p = &Asb[(wm + i * 16 + l16) * LDSS];
      a[i].u4[0] = *(const uint4*)(p + half * 8);        // K 0..7 / 8..15
      a[i].u4[1] = *(const uint4*)(p + 16 + half * 8);   // K 16..23 / 24..31
    }
#pragma unroll
    for (int j = 0; j < 4; ++j) {
      const unsigned short* p = &Bsb[(wn + j * 16 + l16) * LDSS];
      b[j].u4[0] = *(const uint4*)(p + half * 8);
      b[j].u4[1] = *(const uint4*)(p + 16 + half * 8);
    }
#pragma unroll
    for (int i = 0; i < 4; ++i)
#pragma unroll
      for (int j = 0; j < 4; ++j)
        acc[i][j] = __builtin_amdgcn_wmma_f32_16x16x32_bf16(
            false, a[i].v, false, b[j].v, (short)0, acc[i][j], false, false);

    if (more) wait_async_lds();
    __syncthreads();
    cur = nxt;
  }

  // ---- epilogue: bias + activation + bf16 store ----
#pragma unroll
  for (int j = 0; j < 4; ++j) {
    const int gn = blockN + wn + j * 16 + l16;
    if (gn >= Ntot) continue;
    const float bv = (bias != nullptr) ? bias[gn] : 0.0f;
#pragma unroll
    for (int i = 0; i < 4; ++i) {
#pragma unroll
      for (int r = 0; r < 8; ++r) {
        const int gm = blockM + wm + i * 16 + half * 8 + r;
        float v = acc[i][j][r] + bv;
        if (RELU) v = fmaxf(v, 0.0f);
        D[dBatch * z + (size_t)gm * ldd + gn] = (unsigned short)f2bfu(v);
      }
    }
  }
}

// ---- final projection: out[m] = T[m][:] . Wm2 + bm2 ----
__global__ __launch_bounds__(256)
void final_dot_kernel(const unsigned short* __restrict__ T,
                      const float* __restrict__ Wm2,
                      const float* __restrict__ bm2,
                      float* __restrict__ out, int M)
{
  const int m = blockIdx.x * 256 + threadIdx.x;
  if (m >= M) return;
  const unsigned short* t = T + (size_t)m * 64;
  float s = 0.0f;
#pragma unroll
  for (int j = 0; j < 64; ++j) s += bf2f(t[j]) * Wm2[j];
  out[m] = s + bm2[0];
}

extern "C" void kernel_launch(void* const* d_in, const int* in_sizes, int n_in,
                              void* d_out, int out_size, void* d_ws, size_t ws_size,
                              hipStream_t stream) {
  (void)in_sizes; (void)n_in; (void)out_size; (void)ws_size;
  const float* states = (const float*)d_in[0];   // [8,4096,256]
  const float* G      = (const float*)d_in[1];   // [4096,4096]
  const float* W1     = (const float*)d_in[2];   // [256,256]
  const float* b1     = (const float*)d_in[3];
  const float* W2     = (const float*)d_in[4];   // [256,256]
  const float* b2     = (const float*)d_in[5];
  const float* Wm1    = (const float*)d_in[6];   // [256,64]
  const float* bm1    = (const float*)d_in[7];
  const float* Wm2    = (const float*)d_in[8];   // [64,1]
  const float* bm2    = (const float*)d_in[9];
  float* out = (float*)d_out;

  const int Bb = 8, N = 4096, C = 256, HID = 256, MH = 64;
  const int M = Bb * N;                       // 32768
  unsigned short* ws = (unsigned short*)d_ws;
  unsigned short* Gbf = ws;                              // [4096][4096] bf16
  unsigned short* bufA = Gbf + (size_t)N * N;            // [M][HID] bf16
  unsigned short* bufB = bufA + (size_t)M * HID;         // [M][HID] bf16
  unsigned short* Tbuf = bufB + (size_t)M * HID;         // [M][64]  bf16

  const size_t batHN = (size_t)N * HID;
  dim3 blk(256);

  // 0) Gbf = bf16(G) — once, so the hot conv loops have zero-VALU staging
  f32_to_bf16_kernel<<<dim3(((size_t)N * N) / (256 * 8)), blk, 0, stream>>>(G, Gbf);

  // 1) X1 = states @ W1 + b1            (f32 x f32 -> bf16)
  gemm_bias_act_bf16out<false, false, false>
      <<<dim3(M / BLK_M, HID / BLK_N, 1), blk, 0, stream>>>(
          states, C, 0, W1, HID, 0, b1, bufA, HID, 0, HID, C);

  // 2) H1 = relu(G @ X1[b])  per batch  (bf16 x bf16 -> bf16, async staging)
  gemm_bias_act_bf16out<true, true, true>
      <<<dim3(N / BLK_M, HID / BLK_N, Bb), blk, 0, stream>>>(
          Gbf, N, 0, bufA, HID, batHN, nullptr, bufB, HID, batHN, HID, N);

  // 3) X2 = H1 @ W2 + b2                (bf16 x f32 -> bf16)
  gemm_bias_act_bf16out<true, false, false>
      <<<dim3(M / BLK_M, HID / BLK_N, 1), blk, 0, stream>>>(
          bufB, HID, 0, W2, HID, 0, b2, bufA, HID, 0, HID, HID);

  // 4) H2 = relu(G @ X2[b])  per batch
  gemm_bias_act_bf16out<true, true, true>
      <<<dim3(N / BLK_M, HID / BLK_N, Bb), blk, 0, stream>>>(
          Gbf, N, 0, bufA, HID, batHN, nullptr, bufB, HID, batHN, HID, N);

  // 5) T = relu(H2 @ Wm1 + bm1)         (N=64, boundary-guarded)
  gemm_bias_act_bf16out<true, false, true>
      <<<dim3(M / BLK_M, 1, 1), blk, 0, stream>>>(
          bufB, HID, 0, Wm1, MH, 0, bm1, Tbuf, MH, 0, MH, HID);

  // 6) out = T @ Wm2 + bm2
  final_dot_kernel<<<dim3(M / 256), blk, 0, stream>>>(Tbuf, Wm2, bm2, out, M);
}
